// SupervisedGraphSage_75204877353219
// MI455X (gfx1250) — matrix operand order repaired and checked by reading, not compiled, not run on hardware
//
#include <hip/hip_runtime.h>
#include <hip/hip_bf16.h>

#define N_NODES 100000
#define N_FEATS 512
#define HIDDEN  64
#define NSAMP   5
#define NCLS    16
#define BATCH   16384

typedef __attribute__((ext_vector_type(16))) __bf16 v16bf;
typedef __attribute__((ext_vector_type(8)))  __bf16 v8bf;
typedef __attribute__((ext_vector_type(8)))  float  v8f;
typedef __attribute__((ext_vector_type(4)))  float  v4f;

// ---------------------------------------------------------------------------
// Kernel 0: convert weights f32 -> bf16 into workspace
// ---------------------------------------------------------------------------
__global__ void cvt_weights_kernel(const float* __restrict__ w1,
                                   const float* __restrict__ w2,
                                   const float* __restrict__ wc,
                                   __bf16* __restrict__ w1b,
                                   __bf16* __restrict__ w2b,
                                   __bf16* __restrict__ wcb) {
    int i = blockIdx.x * 256 + threadIdx.x;
    if (i < HIDDEN * 2 * N_FEATS) w1b[i] = (__bf16)w1[i];   // 65536
    if (i < HIDDEN * 2 * HIDDEN)  w2b[i] = (__bf16)w2[i];   // 8192
    if (i < NCLS * HIDDEN)        wcb[i] = (__bf16)wc[i];   // 1024
}

// Build a 16x32 bf16 A-fragment from a contiguous f32 row segment.
// ISA A layout (16-bit, 16x32): element i<8 -> K = o+i, i>=8 -> K = 16+o+(i-8),
// where o = 8*(lane>=16). `p` points at the K-chunk base of this lane's row.
__device__ __forceinline__ v16bf a_from_f32(const float* p, int o) {
    v4f f0 = *(const v4f*)(p + o);
    v4f f1 = *(const v4f*)(p + o + 4);
    v4f f2 = *(const v4f*)(p + 16 + o);
    v4f f3 = *(const v4f*)(p + 16 + o + 4);
    v16bf a;
#pragma unroll
    for (int i = 0; i < 4; ++i) {
        a[i]      = (__bf16)f0[i];
        a[i + 4]  = (__bf16)f1[i];
        a[i + 8]  = (__bf16)f2[i];
        a[i + 12] = (__bf16)f3[i];
    }
    return a;
}

// ---------------------------------------------------------------------------
// Kernel 1: layer 1 over all nodes.
// h1[v] = relu( [feat[v] | mean_s feat[nbr[v][s]]] @ w1^T ), stored bf16 [N,64]
// One wave handles a 16-node tile. 4 waves / block.
// ---------------------------------------------------------------------------
__global__ void __launch_bounds__(128)
sage_layer1_kernel(const float* __restrict__ feats,
                   const __bf16* __restrict__ w1b,   // [64][1024] bf16
                   const int* __restrict__ neigh_idx,
                   __bf16* __restrict__ h1) {        // [N][64] bf16
    const int wave = threadIdx.x >> 5;
    const int lane = threadIdx.x & 31;
    const int tile = blockIdx.x * 4 + wave;
    if (tile * 16 >= N_NODES) return;            // wave-uniform exit

    const int laneLo = lane & 15;
    const int hi     = lane >> 4;
    const int o      = hi * 8;

    const int node_m = tile * 16 + laneLo;       // row this lane feeds into A
    const size_t rowSelf = (size_t)node_m * N_FEATS;
    size_t rowN[NSAMP];
#pragma unroll
    for (int s = 0; s < NSAMP; ++s)
        rowN[s] = (size_t)neigh_idx[node_m * NSAMP + s] * N_FEATS;

    v8f acc[4] = {};

    // K = 0..511 : self features; K = 512..1023 : neighbor mean
    for (int kc = 0; kc < 32; ++kc) {
        v16bf a;
        if (kc < 16) {
            a = a_from_f32(feats + rowSelf + kc * 32, o);
        } else {
            const int kb = (kc - 16) * 32;
            v4f s0 = {0,0,0,0}, s1 = {0,0,0,0}, s2 = {0,0,0,0}, s3 = {0,0,0,0};
#pragma unroll
            for (int s = 0; s < NSAMP; ++s) {
                const float* p = feats + rowN[s] + kb;
                s0 += *(const v4f*)(p + o);
                s1 += *(const v4f*)(p + o + 4);
                s2 += *(const v4f*)(p + 16 + o);
                s3 += *(const v4f*)(p + 16 + o + 4);
            }
            const float inv = 1.0f / (float)NSAMP;
#pragma unroll
            for (int i = 0; i < 4; ++i) {
                a[i]      = (__bf16)(s0[i] * inv);
                a[i + 4]  = (__bf16)(s1[i] * inv);
                a[i + 8]  = (__bf16)(s2[i] * inv);
                a[i + 12] = (__bf16)(s3[i] * inv);
            }
        }
#pragma unroll
        for (int nt = 0; nt < 4; ++nt) {
            // B layout: lane holds N = nt*16+laneLo, K = kc*32 + o + i (i<8), kc*32 + 16 + o + (i-8) (i>=8)
            const __bf16* pb = w1b + (size_t)(nt * 16 + laneLo) * (2 * N_FEATS) + kc * 32;
            v8bf b0 = *(const v8bf*)(pb + o);
            v8bf b1 = *(const v8bf*)(pb + 16 + o);
            v16bf b = __builtin_shufflevector(b0, b1, 0,1,2,3,4,5,6,7,8,9,10,11,12,13,14,15);
            acc[nt] = __builtin_amdgcn_wmma_f32_16x16x32_bf16(
                false, a, false, b, (short)0, acc[nt], false, false);
        }
    }

    // ReLU + store bf16. C layout: VGPR r holds M = r + 8*hi, N = laneLo.
#pragma unroll
    for (int nt = 0; nt < 4; ++nt) {
#pragma unroll
        for (int r = 0; r < 8; ++r) {
            float v = acc[nt][r];
            v = v > 0.0f ? v : 0.0f;
            const int nodeRow = tile * 16 + r + hi * 8;
            h1[(size_t)nodeRow * HIDDEN + nt * 16 + laneLo] = (__bf16)v;
        }
    }
}

// ---------------------------------------------------------------------------
// Kernel 2: layer 2 + classifier on the batch.
// h2 = relu([h1[n] | mean h1[nbr[n]]] @ w2^T);  out = h2 @ wcls^T  (f32)
// One wave per 16-row batch tile, 4 waves / block, 1024 tiles total.
// ---------------------------------------------------------------------------
__global__ void __launch_bounds__(128)
sage_layer2_kernel(const __bf16* __restrict__ h1,    // [N][64] bf16
                   const __bf16* __restrict__ w2b,   // [64][128] bf16
                   const __bf16* __restrict__ wcb,   // [16][64] bf16
                   const int* __restrict__ nodes,
                   const int* __restrict__ neigh_idx,
                   float* __restrict__ out) {        // [B][16] f32
    __shared__ __bf16 h2s[4][16][HIDDEN];            // per-wave staging, 8 KB

    const int wave = threadIdx.x >> 5;
    const int lane = threadIdx.x & 31;
    const int tile = blockIdx.x * 4 + wave;          // always < 1024

    const int laneLo = lane & 15;
    const int hi     = lane >> 4;
    const int o      = hi * 8;

    const int node = nodes[tile * 16 + laneLo];
    const size_t rowSelf = (size_t)node * HIDDEN;
    size_t rowN[NSAMP];
#pragma unroll
    for (int s = 0; s < NSAMP; ++s)
        rowN[s] = (size_t)neigh_idx[node * NSAMP + s] * HIDDEN;

    v8f acc[4] = {};
    // K = 0..63 self (kc 0,1); K = 64..127 neighbor mean (kc 2,3)
#pragma unroll
    for (int kc = 0; kc < 4; ++kc) {
        v16bf a;
        if (kc < 2) {
            const __bf16* p = h1 + rowSelf + kc * 32;
            v8bf f0 = *(const v8bf*)(p + o);
            v8bf f1 = *(const v8bf*)(p + 16 + o);
#pragma unroll
            for (int i = 0; i < 8; ++i) { a[i] = f0[i]; a[i + 8] = f1[i]; }
        } else {
            const int kb = (kc - 2) * 32;
            float t0[8] = {}, t1[8] = {};
#pragma unroll
            for (int s = 0; s < NSAMP; ++s) {
                const __bf16* p = h1 + rowN[s] + kb;
                v8bf g0 = *(const v8bf*)(p + o);
                v8bf g1 = *(const v8bf*)(p + 16 + o);
#pragma unroll
                for (int i = 0; i < 8; ++i) { t0[i] += (float)g0[i]; t1[i] += (float)g1[i]; }
            }
            const float inv = 1.0f / (float)NSAMP;
#pragma unroll
            for (int i = 0; i < 8; ++i) {
                a[i]     = (__bf16)(t0[i] * inv);
                a[i + 8] = (__bf16)(t1[i] * inv);
            }
        }
#pragma unroll
        for (int nt = 0; nt < 4; ++nt) {
            const __bf16* pb = w2b + (size_t)(nt * 16 + laneLo) * (2 * HIDDEN) + kc * 32;
            v8bf b0 = *(const v8bf*)(pb + o);
            v8bf b1 = *(const v8bf*)(pb + 16 + o);
            v16bf b = __builtin_shufflevector(b0, b1, 0,1,2,3,4,5,6,7,8,9,10,11,12,13,14,15);
            acc[nt] = __builtin_amdgcn_wmma_f32_16x16x32_bf16(
                false, a, false, b, (short)0, acc[nt], false, false);
        }
    }

    // ReLU, stage h2 (16x64) to LDS to reshape C-layout -> A-layout
#pragma unroll
    for (int nt = 0; nt < 4; ++nt) {
#pragma unroll
        for (int r = 0; r < 8; ++r) {
            float v = acc[nt][r];
            v = v > 0.0f ? v : 0.0f;
            h2s[wave][r + hi * 8][nt * 16 + laneLo] = (__bf16)v;
        }
    }
    __syncthreads();   // all 4 waves always reach this point

    // classifier: out = h2 @ wcls^T  (M=16, K=64, N=16)
    v8f accC = {};
#pragma unroll
    for (int kc = 0; kc < 2; ++kc) {
        const __bf16* p = &h2s[wave][laneLo][kc * 32];
        v8bf f0 = *(const v8bf*)(p + o);
        v8bf f1 = *(const v8bf*)(p + 16 + o);
        v16bf a;
#pragma unroll
        for (int i = 0; i < 8; ++i) { a[i] = f0[i]; a[i + 8] = f1[i]; }
        const __bf16* pb = wcb + (size_t)laneLo * HIDDEN + kc * 32;
        v8bf b0 = *(const v8bf*)(pb + o);
        v8bf b1 = *(const v8bf*)(pb + 16 + o);
        v16bf b = __builtin_shufflevector(b0, b1, 0,1,2,3,4,5,6,7,8,9,10,11,12,13,14,15);
        accC = __builtin_amdgcn_wmma_f32_16x16x32_bf16(
            false, a, false, b, (short)0, accC, false, false);
    }
#pragma unroll
    for (int r = 0; r < 8; ++r) {
        const int row = tile * 16 + r + hi * 8;
        out[(size_t)row * NCLS + laneLo] = accC[r];
    }
}

// ---------------------------------------------------------------------------
extern "C" void kernel_launch(void* const* d_in, const int* in_sizes, int n_in,
                              void* d_out, int out_size, void* d_ws, size_t ws_size,
                              hipStream_t stream) {
    (void)in_sizes; (void)n_in; (void)out_size; (void)ws_size;

    const float* feats     = (const float*)d_in[0];  // [100000,512]
    const float* w1        = (const float*)d_in[1];  // [64,1024]
    const float* w2        = (const float*)d_in[2];  // [64,128]
    const float* wc        = (const float*)d_in[3];  // [16,64]
    const int*   nodes     = (const int*)d_in[4];    // [16384]
    const int*   neigh_idx = (const int*)d_in[5];    // [100000,5]
    float*       out       = (float*)d_out;          // [16384,16]

    // workspace layout
    char* ws = (char*)d_ws;
    __bf16* h1  = (__bf16*)ws;                                   // 12,800,000 B
    __bf16* w1b = (__bf16*)(ws + 12800000);                      // 131,072 B
    __bf16* w2b = (__bf16*)(ws + 12800000 + 131072);             // 16,384 B
    __bf16* wcb = (__bf16*)(ws + 12800000 + 131072 + 16384);     // 2,048 B

    // 0) weight conversion (65536 elems dominates -> 256 blocks x 256)
    cvt_weights_kernel<<<256, 256, 0, stream>>>(w1, w2, wc, w1b, w2b, wcb);

    // 1) layer 1 over all nodes: 6250 tiles of 16, 4 waves/block
    const int tiles1  = (N_NODES + 15) / 16;         // 6250
    const int blocks1 = (tiles1 + 3) / 4;            // 1563
    sage_layer1_kernel<<<blocks1, 128, 0, stream>>>(feats, w1b, neigh_idx, h1);

    // 2) layer 2 + classifier: 1024 tiles of 16, 4 waves/block -> 256 blocks
    sage_layer2_kernel<<<256, 128, 0, stream>>>(h1, w2b, wcb, nodes, neigh_idx, out);
}
